// AudioEncoder_77472620085868
// MI455X (gfx1250) — matrix operand order repaired and checked by reading, not compiled, not run on hardware
//
#include <hip/hip_runtime.h>

#define DEV __device__ __forceinline__

typedef __attribute__((ext_vector_type(16))) __bf16 v16bf;
typedef __attribute__((ext_vector_type(8)))  float  v8f;

struct U4 { unsigned int x, y, z, w; };          // POD 16-byte chunk
union  FragU { v16bf v; U4 q[2]; unsigned short us[16]; };
union  AccU  { v8f v; float f[8]; };

constexpr int T_  = 2048;
constexpr int B_  = 16;
constexpr int H_  = 256;
constexpr int P_  = 128;
constexpr int G4_ = 1024;   // 4*H

// ---- LDS layout for the scan kernel (dynamic shared memory, offsets are the
//      real LDS byte offsets since there is no static __shared__) ----
constexpr int      HS       = 136;                      // padded h row stride (bf16 elems)
constexpr int      SS       = 264;                      // padded s row stride (bf16 elems)
constexpr unsigned WHH_OFF  = 0;                        // W_hh bf16 resident: 262144 B
constexpr unsigned H_OFF    = 262144;                   // h state bf16: 16*136*2 = 4352 B
constexpr unsigned S_OFF    = 266496;                   // s bf16: 16*264*2 = 8448 B
constexpr unsigned RED_OFF  = 274944;                   // split-K reduce: 8*256*4 = 8192 B
constexpr unsigned XP_OFF   = 283136;                   // xp(t) slice f16: 16*1024*2 = 32768 B
constexpr unsigned SMEM_BYTES = XP_OFF + 32768;         // 315904 < 320 KB

DEV unsigned short f2bfr(float f) {
    union { float f; unsigned int u; } c; c.f = f;
    unsigned int u = c.u;
    u += 0x7FFFu + ((u >> 16) & 1u);                    // round-to-nearest-even
    return (unsigned short)(u >> 16);
}
DEV float fsig(float x)  { return 1.0f / (1.0f + __expf(-x)); }
DEV float ftanh(float x) {
    x = fminf(fmaxf(x, -15.0f), 15.0f);
    float e = __expf(2.0f * x);
    return (e - 1.0f) / (e + 1.0f);
}

// CDNA5 async global->LDS (ASYNCcnt-tracked DMA; 16 B per lane per op).
DEV void async_ld_b128(unsigned lds_byte_off, unsigned gl_byte_off, const void* base) {
    asm volatile("global_load_async_to_lds_b128 %0, %1, %2"
                 :: "v"(lds_byte_off), "v"(gl_byte_off), "s"(base) : "memory");
}
DEV void wait_async0() { asm volatile("s_wait_asynccnt 0" ::: "memory"); }

// =====================================================================
// Kernel 0: f32 -> bf16 raw conversion (weights / activations), n % 8 == 0
// =====================================================================
__global__ void cvt_bf16_kernel(const float* __restrict__ src,
                                unsigned short* __restrict__ dst, int n) {
    int i = (blockIdx.x * blockDim.x + threadIdx.x) * 8;
    if (i < n) {
        #pragma unroll
        for (int c = 0; c < 8; ++c) dst[i + c] = f2bfr(src[i + c]);
    }
}

// =====================================================================
// Kernel 1: gate precompute  xp[dir][t][b][n] = X @ W_ihˆT + b_ih + b_hh
//   X (bf16, (B,T,D)), W_ih (bf16, (4H,D)), output f16 in (T,B,4H).
//   grid = (4H/128, T, 2 dirs); block = 256 (8 waves, one 16-wide N tile each)
// =====================================================================
__global__ void lstm_gates_kernel(const unsigned short* __restrict__ Xb,
                                  const unsigned short* __restrict__ Wb0,
                                  const unsigned short* __restrict__ Wb1,
                                  const float* __restrict__ bih0,
                                  const float* __restrict__ bhh0,
                                  const float* __restrict__ bih1,
                                  const float* __restrict__ bhh1,
                                  _Float16* __restrict__ xp0,
                                  _Float16* __restrict__ xp1,
                                  int D)
{
    const int dir = blockIdx.z;
    const unsigned short* W  = dir ? Wb1  : Wb0;
    const float*          bi = dir ? bih1 : bih0;
    const float*          bh = dir ? bhh1 : bhh0;
    _Float16*             xp = dir ? xp1  : xp0;

    const int lane = threadIdx.x & 31;
    const int wave = threadIdx.x >> 5;
    const int l15  = lane & 15;
    const int hi   = lane >> 4;                  // wave half
    const int t    = blockIdx.y;
    const int n0   = blockIdx.x * 128 + wave * 16;

    const unsigned short* arow = Xb + ((size_t)l15 * T_ + t) * D + hi * 16;  // A: m = batch
    const unsigned short* brow = W + (size_t)(n0 + l15) * D + hi * 16;       // B[k][n] = W[n][k]

    AccU acc; acc.v = {};
    const int nK = D >> 5;
    for (int kf = 0; kf < nK; ++kf) {
        FragU a, b;
        const unsigned short* pa = arow + kf * 32;
        a.q[0] = *(const U4*)pa;                 // K kb..kb+7
        a.q[1] = *(const U4*)(pa + 8);           // K kb+8..kb+15
        const unsigned short* pb = brow + kf * 32;
        b.q[0] = *(const U4*)pb;                 // 16 contiguous K
        b.q[1] = *(const U4*)(pb + 8);
        acc.v = __builtin_amdgcn_wmma_f32_16x16x32_bf16(false, a.v, false, b.v,
                                                        (short)0, acc.v, false, false);
    }

    const float bias = bi[n0 + l15] + bh[n0 + l15];
    #pragma unroll
    for (int r = 0; r < 8; ++r) {                // D elem (r,lane) -> (m = r+8*hi, n = n0+l15)
        const int b = r + hi * 8;
        xp[((size_t)t * B_ + b) * G4_ + n0 + l15] = (_Float16)(acc.f[r] + bias);
    }
}

// =====================================================================
// Kernel 2: recurrent scan, one persistent workgroup per direction.
//   512 threads = 16 waves.  W_hh resident in LDS (bf16, async-DMA'd in),
//   W_hr frags in registers, c in f32 registers, per-step xp slice
//   prefetched via global_load_async_to_lds_b128 overlapping the WMMAs.
//   Wave w owns hh-tile w with all four gate N-tiles -> gates stay in regs.
// =====================================================================
__global__ void lstm_scan_kernel(const _Float16* __restrict__ xph0,
                                 const _Float16* __restrict__ xph1,
                                 const unsigned short* __restrict__ WhhB0,
                                 const unsigned short* __restrict__ WhhB1,
                                 const unsigned short* __restrict__ WhrB0,
                                 const unsigned short* __restrict__ WhrB1,
                                 unsigned short* __restrict__ outb,  // (B,T,2P) bf16
                                 float* __restrict__ outf,           // (B,T,2P) f32
                                 int write_f32)
{
    extern __shared__ char smem[];
    unsigned short* whh_s = (unsigned short*)(smem + WHH_OFF);
    unsigned short* h_s   = (unsigned short*)(smem + H_OFF);
    unsigned short* s_s   = (unsigned short*)(smem + S_OFF);
    float*          red   = (float*)(smem + RED_OFF);
    const _Float16* xp_l  = (const _Float16*)(smem + XP_OFF);

    const int dir = blockIdx.x;
    const _Float16*       xph = dir ? xph1  : xph0;
    const unsigned short* Whh = dir ? WhhB1 : WhhB0;
    const unsigned short* Whr = dir ? WhrB1 : WhrB0;

    const int tid  = threadIdx.x;
    const int lane = tid & 31;
    const int w    = tid >> 5;          // wave 0..15
    const int l15  = lane & 15;
    const int hi   = lane >> 4;

    // ---- async DMA W_hh bf16 (4H x P) into LDS: 512 B per thread ----
    for (int j = 0; j < 32; ++j) {
        const unsigned off = (unsigned)tid * 512 + (unsigned)j * 16;
        async_ld_b128(WHH_OFF + off, off, (const void*)Whh);
    }
    // h0 = 0
    for (int i = tid; i < B_ * HS; i += 512) h_s[i] = 0;

    // ---- W_hr bf16 fragment preload: 8 N tiles x 8 K frags over 16 waves ----
    const int ntile2 = w & 7;           // output N tile (of 128 cols)
    const int khalf  = w >> 3;          // K half (frags 0..3 vs 4..7)
    v16bf whr_frag[4];
    {
        const unsigned short* prow = Whr + (size_t)(ntile2 * 16 + l15) * H_;
        #pragma unroll
        for (int j = 0; j < 4; ++j) {
            const int kk = (khalf * 4 + j) * 32 + hi * 16;
            FragU b;
            b.q[0] = *(const U4*)(prow + kk);
            b.q[1] = *(const U4*)(prow + kk + 8);
            whr_frag[j] = b.v;
        }
    }

    float c_reg[8];
    #pragma unroll
    for (int r = 0; r < 8; ++r) c_reg[r] = 0.0f;

    wait_async0();
    __syncthreads();

    const int hh0 = w * 16 + l15;       // this thread's hidden column

    for (int tt = 0; tt < T_; ++tt) {
        const int t = dir ? (T_ - 1 - tt) : tt;

        // ---- issue async prefetch of xp(t) slice (16x1024 f16 = 32 KB) ----
        {
            const unsigned off = (unsigned)tid * 64;
            const void* base = (const void*)(xph + (size_t)t * B_ * G4_);
            async_ld_b128(XP_OFF + off,      off,      base);
            async_ld_b128(XP_OFF + off + 16, off + 16, base);
            async_ld_b128(XP_OFF + off + 32, off + 32, base);
            async_ld_b128(XP_OFF + off + 48, off + 48, base);
        }

        // ---- stage 1 (overlaps DMA): gates = h @ W_hhˆT, wave-local regs ----
        FragU af[4];
        #pragma unroll
        for (int kf = 0; kf < 4; ++kf) {
            const unsigned short* p = h_s + l15 * HS + kf * 32 + hi * 16;
            af[kf].q[0] = *(const U4*)p;
            af[kf].q[1] = *(const U4*)(p + 8);
        }
        AccU acc[4];                        // gate g -> N tile g*256 + w*16
        #pragma unroll
        for (int g = 0; g < 4; ++g) {
            acc[g].v = {};
            const int n0 = g * 256 + w * 16;
            #pragma unroll
            for (int kf = 0; kf < 4; ++kf) {
                const unsigned short* p = whh_s + (size_t)(n0 + l15) * P_ + kf * 32 + hi * 16;
                FragU b;
                b.q[0] = *(const U4*)p;
                b.q[1] = *(const U4*)(p + 8);
                acc[g].v = __builtin_amdgcn_wmma_f32_16x16x32_bf16(false, af[kf].v, false, b.v,
                                                                   (short)0, acc[g].v, false, false);
            }
        }

        wait_async0();
        __syncthreads();                    // xp slice published

        // ---- elementwise cell update, all in registers ----
        #pragma unroll
        for (int r = 0; r < 8; ++r) {
            const int bb = r + hi * 8;
            const float iv = acc[0].f[r] + (float)xp_l[bb * G4_ +       hh0];
            const float fv = acc[1].f[r] + (float)xp_l[bb * G4_ + 256 + hh0];
            const float gv = acc[2].f[r] + (float)xp_l[bb * G4_ + 512 + hh0];
            const float ov = acc[3].f[r] + (float)xp_l[bb * G4_ + 768 + hh0];
            const float c  = fsig(fv) * c_reg[r] + fsig(iv) * ftanh(gv);
            c_reg[r] = c;
            s_s[bb * SS + hh0] = f2bfr(fsig(ov) * ftanh(c));
        }
        __syncthreads();                    // s published

        // ---- stage 2: h = s @ W_hrˆT (16x128), split-K over wave pairs ----
        AccU acc2; acc2.v = {};
        #pragma unroll
        for (int j = 0; j < 4; ++j) {
            const int kf = khalf * 4 + j;
            const unsigned short* p = s_s + l15 * SS + kf * 32 + hi * 16;
            FragU a;
            a.q[0] = *(const U4*)p;
            a.q[1] = *(const U4*)(p + 8);
            acc2.v = __builtin_amdgcn_wmma_f32_16x16x32_bf16(false, a.v, false, whr_frag[j],
                                                             (short)0, acc2.v, false, false);
        }
        if (khalf == 1) {
            #pragma unroll
            for (int r = 0; r < 8; ++r) red[ntile2 * 256 + r * 32 + lane] = acc2.f[r];
        }
        __syncthreads();                    // partials published
        if (khalf == 0) {
            #pragma unroll
            for (int r = 0; r < 8; ++r) {
                const float v  = acc2.f[r] + red[ntile2 * 256 + r * 32 + lane];
                const int   bb = r + hi * 8;
                const int   n  = ntile2 * 16 + l15;
                h_s[bb * HS + n] = f2bfr(v);
                const size_t oi = ((size_t)bb * T_ + t) * (2 * P_) + dir * P_ + n;
                outb[oi] = f2bfr(v);
                if (write_f32) outf[oi] = v;
            }
        }
        __syncthreads();                    // h published; red/xp reads retired
    }
}

// =====================================================================
extern "C" void kernel_launch(void* const* d_in, const int* in_sizes, int n_in,
                              void* d_out, int out_size, void* d_ws, size_t ws_size,
                              hipStream_t stream) {
    (void)in_sizes; (void)n_in; (void)out_size; (void)ws_size;
    const float* x = (const float*)d_in[0];
    auto prm = [&](int l, int d, int j) -> const float* {
        // per (layer,dir): 0=W_ih 1=W_hh 2=b_ih 3=b_hh 4=W_hr
        return (const float*)d_in[1 + (l * 2 + d) * 5 + j];
    };
    auto cvt = [&](const float* src, unsigned short* dst, int n) {
        cvt_bf16_kernel<<<dim3((n / 8 + 255) / 256), 256, 0, stream>>>(src, dst, n);
    };

    char* wp = (char*)d_ws;
    size_t off = 0;
    auto carve = [&](size_t bytes) -> char* {
        char* p = wp + off; off += (bytes + 255) & ~(size_t)255; return p;
    };
    _Float16*       xp0   = (_Float16*)carve((size_t)T_ * B_ * G4_ * 2);
    _Float16*       xp1   = (_Float16*)carve((size_t)T_ * B_ * G4_ * 2);
    unsigned short* xbf   = (unsigned short*)carve((size_t)B_ * T_ * 1024 * 2);
    unsigned short* obf0  = (unsigned short*)carve((size_t)B_ * T_ * 256 * 2);
    unsigned short* obf1  = (unsigned short*)carve((size_t)B_ * T_ * 256 * 2);
    unsigned short* wih_b[2] = { (unsigned short*)carve(1024 * 1024 * 2),
                                 (unsigned short*)carve(1024 * 1024 * 2) };
    unsigned short* whh_b[2] = { (unsigned short*)carve(1024 * 128 * 2),
                                 (unsigned short*)carve(1024 * 128 * 2) };
    unsigned short* whr_b[2] = { (unsigned short*)carve(128 * 256 * 2),
                                 (unsigned short*)carve(128 * 256 * 2) };

    cvt(x, xbf, B_ * T_ * 1024);

    for (int l = 0; l < 4; ++l) {
        const int D = (l == 0) ? 1024 : 256;
        for (int d = 0; d < 2; ++d) {
            cvt(prm(l, d, 0), wih_b[d], G4_ * D);
            cvt(prm(l, d, 1), whh_b[d], G4_ * P_);
            cvt(prm(l, d, 4), whr_b[d], P_ * H_);
        }
        const unsigned short* in =
            (l == 0) ? xbf : ((l & 1) ? obf0 : obf1);           // l1<-obf0, l2<-obf1, l3<-obf0
        unsigned short* outbf = (l & 1) ? obf1 : obf0;          // l0->obf0, l1->obf1, l2->obf0, l3->obf1

        lstm_gates_kernel<<<dim3(G4_ / 128, T_, 2), 256, 0, stream>>>(
            in, wih_b[0], wih_b[1],
            prm(l, 0, 2), prm(l, 0, 3), prm(l, 1, 2), prm(l, 1, 3),
            xp0, xp1, D);

        lstm_scan_kernel<<<dim3(2), 512, SMEM_BYTES, stream>>>(
            xp0, xp1, whh_b[0], whh_b[1], whr_b[0], whr_b[1],
            outbf, (float*)d_out, (l == 3) ? 1 : 0);
    }
}
